// PrimalLinear_85392539779587
// MI455X (gfx1250) — compile-verified, matrix operand order checked by reading, not run on hardware
//
#include <hip/hip_runtime.h>

typedef __attribute__((ext_vector_type(16))) _Float16 v16h;
typedef __attribute__((ext_vector_type(8)))  _Float16 v8h;
typedef __attribute__((ext_vector_type(4)))  _Float16 v4h;
typedef __attribute__((ext_vector_type(8)))  float    v8f;
typedef __attribute__((ext_vector_type(4)))  float    v4f;
typedef __attribute__((ext_vector_type(4)))  int      v4i;

#define M_DIM 1024
#define N_DIM 8192
#define K_DIM 4096
#define LDT   72   // LDS tile row stride in halves (64 + 8 pad -> conflict-free b128 reads)

#if __has_builtin(__builtin_amdgcn_global_load_async_to_lds_b128)
#define HAVE_ASYNC_LDS 1
#else
#define HAVE_ASYNC_LDS 0
#endif

#define AS_GLOBAL __attribute__((address_space(1)))
#define AS_LDS    __attribute__((address_space(3)))

__device__ __forceinline__ void wait_async0() {
#if __has_builtin(__builtin_amdgcn_s_wait_asynccnt)
    __builtin_amdgcn_s_wait_asynccnt(0);
#else
    asm volatile("s_wait_asynccnt 0" ::: "memory");
#endif
}

#if HAVE_ASYNC_LDS
// Async global -> LDS b128 copy (per-lane 16B), tracked by ASYNCcnt.
// Builtin signature (from probe diagnostic): param0 = AS(1) int4*, param1 = AS(3) int4*,
// then immediate offset and cpol.
__device__ __forceinline__ void async_cp_b128(const _Float16* gsrc, _Float16* ldst) {
    const AS_GLOBAL v4i* gc = (const AS_GLOBAL v4i*)gsrc;
    AS_GLOBAL v4i* g = const_cast<AS_GLOBAL v4i*>(gc);
    AS_LDS v4i* l = (AS_LDS v4i*)ldst;
    __builtin_amdgcn_global_load_async_to_lds_b128(g, l, 0, 0);
}
#endif

// ---------------------------------------------------------------------------
// LUT nearest-neighbor quantizer (first-min tie-break via strict <)
// ---------------------------------------------------------------------------
__device__ __forceinline__ float primal_quantize1(float ws) {
    const float lut[16] = {0.0f, 1.0f, -1.0f, 0.5f, -0.5f, 0.333333f, -0.333333f,
                           0.2f, -0.2f, 0.142857f, -0.142857f, 0.090909f, -0.090909f,
                           0.076923f, -0.076923f, 0.0f};
    float best = __builtin_fabsf(ws - lut[0]);
    float q    = lut[0];
#pragma unroll
    for (int i = 1; i < 16; ++i) {
        float d = __builtin_fabsf(ws - lut[i]);
        if (d < best) { best = d; q = lut[i]; }
    }
    return q;
}

__global__ __launch_bounds__(256)
void quant_w_kernel(const float* __restrict__ w, const float* __restrict__ scale,
                    _Float16* __restrict__ wq, int n4) {
    int i = blockIdx.x * blockDim.x + threadIdx.x;
    if (i >= n4) return;
    const float s   = scale[0];
    const float inv = 1.0f / s;
    v4f v = ((const v4f*)w)[i];
    v4h h;
#pragma unroll
    for (int j = 0; j < 4; ++j)
        h[j] = (_Float16)(primal_quantize1(v[j] * inv) * s);
    ((v4h*)wq)[i] = h;
}

__global__ __launch_bounds__(256)
void cvt_x_kernel(const float* __restrict__ x, _Float16* __restrict__ xh, int n4) {
    int i = blockIdx.x * blockDim.x + threadIdx.x;
    if (i >= n4) return;
    v4f v = ((const v4f*)x)[i];
    v4h h;
#pragma unroll
    for (int j = 0; j < 4; ++j) h[j] = (_Float16)v[j];
    ((v4h*)xh)[i] = h;
}

// ---------------------------------------------------------------------------
// WMMA GEMM: C[M,N] = A[M,K] * B[N,K]^T   (f16 in, f32 accumulate)
// Block: 256 threads = 8 waves; block tile 128x128; wave tile 64x32; K-step 64.
// Double-buffered LDS fed by GLOBAL_LOAD_ASYNC_TO_LDS_B128 (ASYNCcnt-tracked).
// ---------------------------------------------------------------------------
__global__ __launch_bounds__(256)
void gemm_kernel(const _Float16* __restrict__ A,   // x  in f16, [M, K]
                 const _Float16* __restrict__ B,   // Wq in f16, [N, K]
                 float* __restrict__ C) {
    __shared__ _Float16 lA[2][128 * LDT];
    __shared__ _Float16 lB[2][128 * LDT];

    const int tid  = threadIdx.x;
    const int lane = tid & 31;          // wave32
    const int wave = tid >> 5;          // 0..7
    const int hl   = lane >> 4;         // half-wave: 0/1
    const int r    = lane & 15;

    const int mBase = blockIdx.y * 128;
    const int nBase = blockIdx.x * 128;
    const int mW    = (wave >> 2) * 64; // wave row offset inside block tile (0/64)
    const int nW    = (wave & 3) * 32;  // wave col offset inside block tile (0..96)

    v8f acc[4][2];
    const v8f vzero = {0.f, 0.f, 0.f, 0.f, 0.f, 0.f, 0.f, 0.f};
#pragma unroll
    for (int mi = 0; mi < 4; ++mi)
#pragma unroll
        for (int ni = 0; ni < 2; ++ni) acc[mi][ni] = vzero;

#if HAVE_ASYNC_LDS
    // Direct global -> LDS async staging (per-lane b128), no VGPR bounce.
    auto issue_stage = [&](int k0, int buf) {
#pragma unroll
        for (int j = 0; j < 4; ++j) {
            int idx = (j * 256 + tid) * 8;   // half-element linear index in 128x64 tile
            int row = idx >> 6;
            int col = idx & 63;
            async_cp_b128(&A[(size_t)(mBase + row) * K_DIM + k0 + col],
                          &lA[buf][row * LDT + col]);
            async_cp_b128(&B[(size_t)(nBase + row) * K_DIM + k0 + col],
                          &lB[buf][row * LDT + col]);
        }
    };
#else
    auto copy_stage = [&](int k0, int buf) {
#pragma unroll
        for (int j = 0; j < 4; ++j) {
            int idx = (j * 256 + tid) * 8;
            int row = idx >> 6;
            int col = idx & 63;
            *(v8h*)&lA[buf][row * LDT + col] =
                *(const v8h*)&A[(size_t)(mBase + row) * K_DIM + k0 + col];
            *(v8h*)&lB[buf][row * LDT + col] =
                *(const v8h*)&B[(size_t)(nBase + row) * K_DIM + k0 + col];
        }
    };
#endif

    auto compute_stage = [&](int buf) {
        const _Float16* __restrict__ pA = lA[buf];
        const _Float16* __restrict__ pB = lB[buf];
#pragma unroll
        for (int kk = 0; kk < 64; kk += 32) {
            // A fragment (16x32, MxK): lane r holds row (mW+mi*16+r);
            // K = kk + hl*8 + [0..7] in VGPR0-3, K = kk + 16 + hl*8 + [0..7] in VGPR4-7.
            v16h aF[4], bF[2];
#pragma unroll
            for (int mi = 0; mi < 4; ++mi) {
                const _Float16* p = &pA[(mW + mi * 16 + r) * LDT + kk + hl * 8];
                v8h lo = *(const v8h*)p;
                v8h hi = *(const v8h*)(p + 16);
                aF[mi] = __builtin_shufflevector(lo, hi, 0, 1, 2, 3, 4, 5, 6, 7,
                                                 8, 9, 10, 11, 12, 13, 14, 15);
            }
            // B fragment (32x16, KxN): lane r holds column (nW+ni*16+r), same K striping.
#pragma unroll
            for (int ni = 0; ni < 2; ++ni) {
                const _Float16* p = &pB[(nW + ni * 16 + r) * LDT + kk + hl * 8];
                v8h lo = *(const v8h*)p;
                v8h hi = *(const v8h*)(p + 16);
                bF[ni] = __builtin_shufflevector(lo, hi, 0, 1, 2, 3, 4, 5, 6, 7,
                                                 8, 9, 10, 11, 12, 13, 14, 15);
            }
#pragma unroll
            for (int mi = 0; mi < 4; ++mi)
#pragma unroll
                for (int ni = 0; ni < 2; ++ni)
                    acc[mi][ni] = __builtin_amdgcn_wmma_f32_16x16x32_f16(
                        /*neg_a=*/false, aF[mi], /*neg_b=*/false, bF[ni],
                        /*c_mod=*/(short)0, acc[mi][ni],
                        /*reuse_a=*/false, /*reuse_b=*/false);
        }
    };

#if HAVE_ASYNC_LDS
    issue_stage(0, 0);
    int buf = 0;
    for (int k0 = 0; k0 < K_DIM; k0 += 64, buf ^= 1) {
        wait_async0();       // own wave's stage-k0 loads have landed in LDS
        __syncthreads();     // everyone's stage-k0 data visible; prev compute done
        if (k0 + 64 < K_DIM) issue_stage(k0 + 64, buf ^ 1);
        compute_stage(buf);
    }
#else
    for (int k0 = 0; k0 < K_DIM; k0 += 64) {
        __syncthreads();
        copy_stage(k0, 0);
        __syncthreads();
        compute_stage(0);
    }
#endif

    // C/D layout: VGPR g holds M = g + 8*hl (per half-wave), N = lane&15.
#pragma unroll
    for (int mi = 0; mi < 4; ++mi)
#pragma unroll
        for (int ni = 0; ni < 2; ++ni) {
            int col = nBase + nW + ni * 16 + r;
#pragma unroll
            for (int g = 0; g < 8; ++g) {
                int row = mBase + mW + mi * 16 + hl * 8 + g;
                C[(size_t)row * N_DIM + col] = acc[mi][ni][g];
            }
        }
}

// ---------------------------------------------------------------------------
extern "C" void kernel_launch(void* const* d_in, const int* in_sizes, int n_in,
                              void* d_out, int out_size, void* d_ws, size_t ws_size,
                              hipStream_t stream) {
    const float* x     = (const float*)d_in[0];   // [1,1024,4096] f32
    const float* w     = (const float*)d_in[1];   // [8192,4096]   f32
    const float* scale = (const float*)d_in[2];   // scalar        f32
    float* out = (float*)d_out;                   // [1,1024,8192] f32

    _Float16* wq = (_Float16*)d_ws;                                        // 64 MB
    _Float16* xh = (_Float16*)((char*)d_ws + (size_t)N_DIM * K_DIM * 2);   // + 8 MB

    const int nW4 = (N_DIM * K_DIM) / 4;
    quant_w_kernel<<<(nW4 + 255) / 256, 256, 0, stream>>>(w, scale, wq, nW4);

    const int nX4 = (M_DIM * K_DIM) / 4;
    cvt_x_kernel<<<(nX4 + 255) / 256, 256, 0, stream>>>(x, xh, nX4);

    dim3 grid(N_DIM / 128, M_DIM / 128);   // 64 x 8 = 512 blocks
    gemm_kernel<<<grid, 256, 0, stream>>>(xh, wq, out);
}
